// SpUNetNoSkipBase_64785286693642
// MI455X (gfx1250) — compile-verified, matrix operand order checked by reading, not run on hardware
//
#include <hip/hip_runtime.h>

// ---------------------------------------------------------------------------
// SpUNetNoSkip forward for MI455X (gfx1250, wave32).
// Matrix path: V_WMMA_F32_16X16X32_BF16 (bf16 inputs, f32 accumulate).
// Each wave computes NT adjacent 16x16 output tiles with independent
// accumulators: amortizes the gathered A fragment NT x and keeps >=NT
// instructions between dependent WMMAs (hides the WMMA->WMMA RAW hazard).
// ---------------------------------------------------------------------------

typedef __bf16 bf16_t;
typedef __attribute__((ext_vector_type(16))) __bf16 v16bf;
typedef __attribute__((ext_vector_type(8)))  float  v8f;

#define BN_EPS 1e-3f

// ---------------- elementwise / prep kernels ----------------

__global__ void k_zero_f32(float* __restrict__ p, long long n) {
  long long i = (long long)blockIdx.x * blockDim.x + threadIdx.x;
  if (i < n) p[i] = 0.f;
}

__global__ void k_f32_to_bf16(const float* __restrict__ x, bf16_t* __restrict__ y, long long n) {
  long long i = (long long)blockIdx.x * blockDim.x + threadIdx.x;
  if (i < n) y[i] = (bf16_t)x[i];
}

// weights f32 [K][C][D] -> bf16 [K][D][C] (so B-fragment loads are contiguous)
__global__ void k_wt_bf16(const float* __restrict__ w, bf16_t* __restrict__ wt,
                          int K, int C, int D) {
  long long i = (long long)blockIdx.x * blockDim.x + threadIdx.x;
  long long tot = (long long)K * C * D;
  if (i >= tot) return;
  int d = (int)(i % D);
  long long r = i / D;
  int c = (int)(r % C);
  int k = (int)(r / C);
  wt[((long long)k * D + d) * C + c] = (bf16_t)w[i];
}

// conv_in weights f32 [125][4][32] -> bf16 [32][512] with zero K-padding (500->512)
__global__ void k_wt_in(const float* __restrict__ w, bf16_t* __restrict__ wt) {
  int i = blockIdx.x * blockDim.x + threadIdx.x;
  if (i >= 32 * 512) return;
  int kk = i % 512, d = i / 512;
  float v = (kk < 500) ? w[(long long)kk * 32 + d] : 0.f;
  wt[(long long)d * 512 + kk] = (bf16_t)v;
}

// ---------------- WMMA gather-GEMM (submanifold conv) ----------------
// Y[n][D] = sum_k sum_c (mask ? Xb[nbr[k,n]][c] : 0) * W[k][c][d]
// One wave -> NT adjacent 16x16 output tiles; K chunked by 32 (C % 32 == 0).
// A layout : lanes 0-15 row M=l16 / K 0..15, lanes 16-31 same row / K 16..31.
// B layout : lanes 0-15 col N=l16 / K 0..15, lanes 16-31 same col / K 16..31.
// C/D      : lanes 0-15 M 0..7 in v[0..7], lanes 16-31 M 8..15.
template <int NT>
__global__ void k_subm_gemm(const bf16_t* __restrict__ Xb,
                            const bf16_t* __restrict__ Wt,   // [K][D][C] bf16
                            const int* __restrict__ nbr,      // [K][n]
                            const unsigned char* __restrict__ mask,
                            float* __restrict__ Y,            // [n][D]
                            int n, int C, int D, int Koff) {
  const int lane = threadIdx.x;
  const int l16  = lane & 15;
  const int half = lane >> 4;
  const int tm = blockIdx.x * blockDim.y + threadIdx.y;
  if (tm * 16 >= n) return;               // uniform across the wave
  const int m = tm * 16 + l16;            // A row handled by this lane

  // per-lane B column base for each of the NT tiles (k-advance added in loop)
  const bf16_t* wcol[NT];
#pragma unroll
  for (int t = 0; t < NT; ++t) {
    int d = (blockIdx.y * NT + t) * 16 + l16;
    wcol[t] = Wt + (long long)d * C + half * 16;
  }

  const v8f vzero8 = {};
  v8f acc[NT];
#pragma unroll
  for (int t = 0; t < NT; ++t) acc[t] = vzero8;

  const long long kStride = (long long)D * C;
  for (int k = 0; k < Koff; ++k) {
    bool valid = false;
    int r = 0;
    if (m < n) {
      int e = k * n + m;
      valid = (mask[e] != 0);
      r = nbr[e];
    }
    const bool wave_all = (__all(valid ? 1 : 0) != 0);
    const bf16_t* xrow = Xb + (long long)r * C + half * 16;
    const long long wk = (long long)k * kStride;
    for (int c0 = 0; c0 < C; c0 += 32) {
      v16bf a;
      if (wave_all) {
        a = *(const v16bf*)(xrow + c0);           // fast path: no zero-fill
      } else {
        v16bf z = {};
        a = z;
        if (valid) a = *(const v16bf*)(xrow + c0);
      }
#pragma unroll
      for (int t = 0; t < NT; ++t) {
        v16bf b = *(const v16bf*)(wcol[t] + wk + c0);
        acc[t] = __builtin_amdgcn_wmma_f32_16x16x32_bf16(false, a, false, b,
                                                         (short)0, acc[t], false, false);
      }
    }
  }
#pragma unroll
  for (int t = 0; t < NT; ++t) {
    int d = (blockIdx.y * NT + t) * 16 + l16;
#pragma unroll
    for (int rr = 0; rr < 8; ++rr) {
      int row = tm * 16 + half * 8 + rr;
      if (row < n) Y[(long long)row * D + d] = acc[t][rr];
    }
  }
}

// conv_in: Cin=4, 125 offsets -> synthetic K=500 (padded to 512), D=32 (2 tiles).
__global__ void k_subm_gemm_cin4(const float* __restrict__ X,      // [n][4] f32
                                 const bf16_t* __restrict__ Wt,    // [32][512] bf16
                                 const int* __restrict__ nbr,       // [125][n]
                                 const unsigned char* __restrict__ mask,
                                 float* __restrict__ Y,             // [n][32]
                                 int n) {
  const int lane = threadIdx.x;
  const int l16  = lane & 15;
  const int half = lane >> 4;
  const int tm = blockIdx.x * blockDim.y + threadIdx.y;
  if (tm * 16 >= n) return;
  const int m = tm * 16 + l16;

  const bf16_t* wcol0 = Wt + (long long)(l16)      * 512 + half * 16;
  const bf16_t* wcol1 = Wt + (long long)(16 + l16) * 512 + half * 16;

  v8f acc0 = {}, acc1 = {};
  for (int c0 = 0; c0 < 512; c0 += 32) {
    v16bf a = {};
    if (m < n) {
#pragma unroll
      for (int j = 0; j < 16; ++j) {
        int kk = c0 + half * 16 + j;
        if (kk < 500) {
          int k = kk >> 2, ci = kk & 3;
          int e = k * n + m;
          if (mask[e]) a[j] = (bf16_t)X[(long long)nbr[e] * 4 + ci];
        }
      }
    }
    v16bf b0 = *(const v16bf*)(wcol0 + c0);
    v16bf b1 = *(const v16bf*)(wcol1 + c0);
    acc0 = __builtin_amdgcn_wmma_f32_16x16x32_bf16(false, a, false, b0, (short)0, acc0, false, false);
    acc1 = __builtin_amdgcn_wmma_f32_16x16x32_bf16(false, a, false, b1, (short)0, acc1, false, false);
  }
#pragma unroll
  for (int rr = 0; rr < 8; ++rr) {
    int row = tm * 16 + half * 8 + rr;
    if (row < n) {
      Y[(long long)row * 32 + l16]      = acc0[rr];
      Y[(long long)row * 32 + 16 + l16] = acc1[rr];
    }
  }
}

// ---------------- BatchNorm (training-mode batch stats) ----------------

__global__ void k_bn_stats(const float* __restrict__ X, float* __restrict__ stats,
                           int n, int C) {
  int c = threadIdx.x;                    // blockDim.x == C
  float s = 0.f, ss = 0.f;
  for (int r = blockIdx.x; r < n; r += gridDim.x) {
    float v = X[(long long)r * C + c];
    s += v; ss += v * v;
  }
  atomicAdd(&stats[c], s);
  atomicAdd(&stats[C + c], ss);
}

__global__ void k_bn_apply(const float* __restrict__ X, const float* __restrict__ stats,
                           const float* __restrict__ g, const float* __restrict__ b,
                           const float* __restrict__ res, float* __restrict__ Y,
                           int n, int C, int relu) {
  long long i = (long long)blockIdx.x * blockDim.x + threadIdx.x;
  if (i >= (long long)n * C) return;
  int c = (int)(i % C);
  float inv_n = 1.f / (float)n;
  float mu  = stats[c] * inv_n;
  float var = stats[C + c] * inv_n - mu * mu;   // biased variance (matches ref)
  float v = (X[i] - mu) * rsqrtf(var + BN_EPS) * g[c] + b[c];
  if (res) v += res[i];
  if (relu) v = fmaxf(v, 0.f);
  Y[i] = v;
}

// ---------------- stride-2 down conv (scatter-add) ----------------

__global__ void k_down(const float* __restrict__ X, const float* __restrict__ W, // [8][Cin][Cout]
                       const int* __restrict__ parent, const int* __restrict__ off,
                       float* __restrict__ Y, int nf, int Cin, int Cout) {
  long long i = (long long)blockIdx.x * blockDim.x + threadIdx.x;
  if (i >= (long long)nf * Cout) return;
  int nvx = (int)(i / Cout), d = (int)(i % Cout);
  int o = off[nvx], p = parent[nvx];
  const float* w = W + ((long long)o * Cin) * Cout + d;
  const float* x = X + (long long)nvx * Cin;
  float s = 0.f;
  for (int c = 0; c < Cin; ++c) s += x[c] * w[(long long)c * Cout];
  atomicAdd(&Y[(long long)p * Cout + d], s);
}

// ---------------- inverse conv (gather) ----------------

__global__ void k_up(const float* __restrict__ Xc, const float* __restrict__ W, // [8][Cin][Cout]
                     const int* __restrict__ parent, const int* __restrict__ off,
                     float* __restrict__ Y, int nf, int Cin, int Cout) {
  long long i = (long long)blockIdx.x * blockDim.x + threadIdx.x;
  if (i >= (long long)nf * Cout) return;
  int nvx = (int)(i / Cout), d = (int)(i % Cout);
  int o = off[nvx], p = parent[nvx];
  const float* w = W + ((long long)o * Cin) * Cout + d;
  const float* x = Xc + (long long)p * Cin;
  float s = 0.f;
  for (int c = 0; c < Cin; ++c) s += x[c] * w[(long long)c * Cout];
  Y[i] = s;
}

// ---------------- final 96 -> 13 linear ----------------

__global__ void k_final(const float* __restrict__ X, const float* __restrict__ W, // [96][13]
                        const float* __restrict__ B, float* __restrict__ out, int n) {
  long long i = (long long)blockIdx.x * blockDim.x + threadIdx.x;
  if (i >= (long long)n * 13) return;
  int nvx = (int)(i / 13), o = (int)(i % 13);
  float s = B[o];
  const float* x = X + (long long)nvx * 96;
  for (int c = 0; c < 96; ++c) s += x[c] * W[c * 13 + o];
  out[i] = s;
}

// ---------------------------------------------------------------------------
// Host orchestration.  Inputs follow jax tree flatten order (dict keys sorted):
//   feats,
//   params: conv_in{b,g,w}, dec[4][2]{b1,b2,g1,g2,w1,w2}, down[4]{b,g,w},
//           enc[{2,3,4,6}]{b1..w2}, final_b, final_w, up[4]{b,g,w},
//   rulebooks: down[4](parent,off,n_out), subm3[5](nbr,mask), subm5(nbr,mask)
// Per-level voxel counts are derived on the host from in_sizes (nbr = 27*N).
// ---------------------------------------------------------------------------

extern "C" void kernel_launch(void* const* d_in, const int* in_sizes, int n_in,
                              void* d_out, int out_size, void* d_ws, size_t ws_size,
                              hipStream_t stream) {
  (void)n_in; (void)out_size; (void)ws_size;

  int idx = 0;
  const int I_FEATS = idx++;
  const int I_CIN_B = idx++; const int I_CIN_G = idx++; const int I_CIN_W = idx++;
  int DECP[4][2][6];
  for (int s = 0; s < 4; ++s)
    for (int b = 0; b < 2; ++b)
      for (int f = 0; f < 6; ++f) DECP[s][b][f] = idx++;
  int DWN_B[4], DWN_G[4], DWN_W[4];
  for (int s = 0; s < 4; ++s) { DWN_B[s] = idx++; DWN_G[s] = idx++; DWN_W[s] = idx++; }
  const int ENC_L[4] = {2, 3, 4, 6};
  int ENCP[4][6][6];
  for (int s = 0; s < 4; ++s)
    for (int b = 0; b < ENC_L[s]; ++b)
      for (int f = 0; f < 6; ++f) ENCP[s][b][f] = idx++;
  const int I_FIN_B = idx++; const int I_FIN_W = idx++;
  int UP_B[4], UP_G[4], UP_W[4];
  for (int s = 0; s < 4; ++s) { UP_B[s] = idx++; UP_G[s] = idx++; UP_W[s] = idx++; }
  int RB_PAR[4], RB_OFF[4];
  for (int s = 0; s < 4; ++s) { RB_PAR[s] = idx++; RB_OFF[s] = idx++; idx++; /* n_out leaf unused */ }
  int S3_NBR[5], S3_MSK[5];
  for (int l = 0; l < 5; ++l) { S3_NBR[l] = idx++; S3_MSK[l] = idx++; }
  const int I_S5_NBR = idx++; const int I_S5_MSK = idx++;

  int ns[5];
  for (int l = 0; l < 5; ++l) ns[l] = in_sizes[S3_NBR[l]] / 27;
  const int n0 = ns[0];
  const int CH[8] = {32, 64, 128, 256, 256, 128, 96, 96};

  // workspace carve (all 256B aligned)
  char* wp = (char*)d_ws;
  auto carve = [&](size_t bytes) {
    void* p = (void*)wp;
    wp += (bytes + 255) & ~(size_t)255;
    return p;
  };
  float*  X  = (float*)carve((size_t)n0 * 256 * 4);
  float*  H  = (float*)carve((size_t)n0 * 256 * 4);
  float*  T  = (float*)carve((size_t)n0 * 256 * 4);
  bf16_t* Xb = (bf16_t*)carve((size_t)n0 * 256 * 2);
  bf16_t* Wb = (bf16_t*)carve((size_t)27 * 256 * 256 * 2);
  bf16_t* Wi = (bf16_t*)carve((size_t)32 * 512 * 2);
  float*  stats = (float*)carve(512 * 4);

  auto cdivll = [](long long a, long long b) { return (int)((a + b - 1) / b); };

  auto bn = [&](const float* src, int gi, int bi, int n, int C, bool relu,
                const float* res, float* dst) {
    k_zero_f32<<<1, 2 * C, 0, stream>>>(stats, 2 * C);
    k_bn_stats<<<dim3(64), dim3(C), 0, stream>>>(src, stats, n, C);
    long long tot = (long long)n * C;
    k_bn_apply<<<cdivll(tot, 256), 256, 0, stream>>>(
        src, stats, (const float*)d_in[gi], (const float*)d_in[bi], res, dst,
        n, C, relu ? 1 : 0);
  };

  auto subm = [&](const float* src, int wi, int l, int C, int D, float* dst) {
    int n = ns[l];
    k_f32_to_bf16<<<cdivll((long long)n * C, 256), 256, 0, stream>>>(src, Xb, (long long)n * C);
    k_wt_bf16<<<cdivll((long long)27 * C * D, 256), 256, 0, stream>>>(
        (const float*)d_in[wi], Wb, 27, C, D);
    const int tiles_m = (n + 15) / 16;
    dim3 blk(32, 8);
    const int* nb = (const int*)d_in[S3_NBR[l]];
    const unsigned char* mk = (const unsigned char*)d_in[S3_MSK[l]];
    if (D % 64 == 0) {
      dim3 grid((unsigned)cdivll(tiles_m, 8), (unsigned)(D / 64));
      k_subm_gemm<4><<<grid, blk, 0, stream>>>(Xb, Wb, nb, mk, dst, n, C, D, 27);
    } else {
      dim3 grid((unsigned)cdivll(tiles_m, 8), (unsigned)(D / 32));
      k_subm_gemm<2><<<grid, blk, 0, stream>>>(Xb, Wb, nb, mk, dst, n, C, D, 27);
    }
  };

  auto block = [&](int l, int C, const int* p /* b1,b2,g1,g2,w1,w2 */) {
    subm(X, p[4], l, C, C, H);
    bn(H, p[2], p[0], ns[l], C, true, nullptr, H);       // relu(bn(subm(x,w1)))
    subm(H, p[5], l, C, C, T);
    bn(T, p[3], p[1], ns[l], C, true, X, X);             // relu(bn(subm(h,w2)) + x)
  };

  // ---- conv_in: subm k=5, 4 -> 32 ----
  k_wt_in<<<cdivll(32 * 512, 256), 256, 0, stream>>>((const float*)d_in[I_CIN_W], Wi);
  {
    dim3 grid((unsigned)cdivll(((long long)n0 + 15) / 16, 8), 1);
    dim3 blk(32, 8);
    k_subm_gemm_cin4<<<grid, blk, 0, stream>>>(
        (const float*)d_in[I_FEATS], Wi,
        (const int*)d_in[I_S5_NBR], (const unsigned char*)d_in[I_S5_MSK], H, n0);
  }
  bn(H, I_CIN_G, I_CIN_B, n0, 32, true, nullptr, X);

  // ---- encoder ----
  int cin = 32;
  for (int s = 0; s < 4; ++s) {
    int cout = CH[s];
    int nf = ns[s], nc = ns[s + 1];
    k_zero_f32<<<cdivll((long long)nc * cout, 256), 256, 0, stream>>>(H, (long long)nc * cout);
    k_down<<<cdivll((long long)nf * cout, 256), 256, 0, stream>>>(
        X, (const float*)d_in[DWN_W[s]],
        (const int*)d_in[RB_PAR[s]], (const int*)d_in[RB_OFF[s]], H, nf, cin, cout);
    bn(H, DWN_G[s], DWN_B[s], nc, cout, true, nullptr, X);
    for (int b = 0; b < ENC_L[s]; ++b) block(s + 1, cout, ENCP[s][b]);
    cin = cout;
  }

  // ---- decoder (no skip connections) ----
  for (int s = 0; s < 4; ++s) {
    int lvl = 3 - s;
    int cout = CH[4 + s];
    int nf = ns[lvl];
    k_up<<<cdivll((long long)nf * cout, 256), 256, 0, stream>>>(
        X, (const float*)d_in[UP_W[s]],
        (const int*)d_in[RB_PAR[lvl]], (const int*)d_in[RB_OFF[lvl]], H, nf, cin, cout);
    bn(H, UP_G[s], UP_B[s], nf, cout, true, nullptr, X);
    for (int b = 0; b < 2; ++b) block(lvl, cout, DECP[s][b]);
    cin = cout;
  }

  // ---- final 1x1 conv ----
  k_final<<<cdivll((long long)n0 * 13, 256), 256, 0, stream>>>(
      X, (const float*)d_in[I_FIN_W], (const float*)d_in[I_FIN_B], (float*)d_out, n0);
}
